// SJCSNN_86758339379229
// MI455X (gfx1250) — matrix operand order repaired and checked by reading, not compiled, not run on hardware
//
#include <hip/hip_runtime.h>
#include <hip/hip_bf16.h>

// ---------------------------------------------------------------------------
// Spiking CNN forward for MI455X (gfx1250), wave32 + WMMA f16 (f32 accum).
// Spike activations are exactly representable in f16; all big matmuls
// (conv2 implicit GEMM, fc1, fc2) run on v_wmma_f32_16x16x32_f16.
// LDS staging uses GLOBAL_LOAD_ASYNC_TO_LDS_B128 (ASYNCcnt) when the
// toolchain exposes the gfx1250 async builtins; falls back to ds stores.
// ---------------------------------------------------------------------------

typedef __attribute__((ext_vector_type(16))) _Float16 v16h;
typedef __attribute__((ext_vector_type(8)))  _Float16 v8h;
typedef __attribute__((ext_vector_type(8)))  float    v8f;
typedef __attribute__((ext_vector_type(4)))  int      v4i;

#define T_STEPS 8
#define BATCH   32
#define BN_EPS  1e-5f

#if defined(__gfx1250__) && \
    __has_builtin(__builtin_amdgcn_global_load_async_to_lds_b128) && \
    __has_builtin(__builtin_amdgcn_s_wait_asynccnt)
#define USE_ASYNC_STAGE 1
typedef __attribute__((address_space(1))) v4i* gv4i_p;  // global int4*
typedef __attribute__((address_space(3))) v4i* lv4i_p;  // LDS int4*
#define ASYNC_B128(gsrc, ldst)                                            \
  __builtin_amdgcn_global_load_async_to_lds_b128(                         \
      (gv4i_p)(gsrc), (lv4i_p)(ldst), 0, 0)
#define ASYNC_WAIT() __builtin_amdgcn_s_wait_asynccnt(0)
#else
#define USE_ASYNC_STAGE 0
#endif

// Concatenate two 8-half groups into one 16-half WMMA operand register block.
static __device__ __forceinline__ v16h frag_cat(v8h lo, v8h hi) {
  v16h r;
#pragma unroll
  for (int i = 0; i < 8; ++i) { r[i] = lo[i]; r[i + 8] = hi[i]; }
  return r;
}

// ---------------------------------------------------------------------------
// Layout conventions (CDNA5 ISA 7.12.2, wave32):
//  A (16x32 f16): lane holds row m = lane&15; element h of v16h holds
//      k = (h>>3)*16 + (lane>>4)*8 + (h&7)    -> two contiguous 16B LDS loads.
//  B (32x16 f16): lane holds col n = lane&15; element h holds
//      k = (h>>3)*16 + (lane>>4)*8 + (h&7)                   -> one contiguous 32B load/lane
//      (we PREPACK weights into exactly this order).
//  C/D (16x16 f32, 8 VGPRs): row = r + 8*(lane>>4), col = lane&15.
// ---------------------------------------------------------------------------

// ---- weight prep: fold BN2 into conv2, pack into B-fragment order ---------
__global__ __launch_bounds__(256) void pack_conv2_k(
    const float* __restrict__ w, const float* __restrict__ cb,
    const float* __restrict__ g, const float* __restrict__ be,
    const float* __restrict__ mn, const float* __restrict__ vr,
    _Float16* __restrict__ Wp, float* __restrict__ Bf) {
  int pi = blockIdx.x * 256 + threadIdx.x;
  if (pi >= 9 * 8 * 4 * 512) return;
  int h = pi & 15, lane = (pi >> 4) & 31, blk = pi >> 9;
  int kc = blk & 3, nT = (blk >> 2) & 7, tap = blk >> 5;
  int n = nT * 16 + (lane & 15);                 // output channel
  int k = kc * 32 + ((h >> 3) << 4) + ((lane >> 4) << 3) + (h & 7);      // input channel
  float sc = g[n] * rsqrtf(vr[n] + BN_EPS);
  Wp[pi] = (_Float16)(w[(n * 128 + k) * 9 + tap] * sc);
  if (pi < 128) {
    float s2 = g[pi] * rsqrtf(vr[pi] + BN_EPS);
    Bf[pi] = (cb[pi] - mn[pi]) * s2 + be[pi];
  }
}

// ---- generic fc weight pack (w is [N,K] row-major fp32) -------------------
__global__ __launch_bounds__(256) void pack_fc_k(
    const float* __restrict__ w, _Float16* __restrict__ Wp,
    int K, int kBlocks, long long total) {
  long long pi = (long long)blockIdx.x * 256 + threadIdx.x;
  if (pi >= total) return;
  int h = (int)(pi & 15), lane = (int)((pi >> 4) & 31);
  long long blk = pi >> 9;
  int kB = (int)(blk % kBlocks);
  int nT = (int)(blk / kBlocks);
  int n = nT * 16 + (lane & 15);
  int k = kB * 32 + ((h >> 3) << 4) + ((lane >> 4) << 3) + (h & 7);
  Wp[pi] = (_Float16)(w[(long long)n * K + k]);
}

// ---- conv1 + BN1 (tiny: 1 input channel), NHWC f32 output -----------------
__global__ __launch_bounds__(256) void conv1_bn_k(
    const float* __restrict__ X, const float* __restrict__ W,
    const float* __restrict__ B, const float* __restrict__ g,
    const float* __restrict__ be, const float* __restrict__ mn,
    const float* __restrict__ vr, float* __restrict__ H) {
  int idx = blockIdx.x * 256 + threadIdx.x;
  if (idx >= BATCH * 2304 * 128) return;
  int c = idx & 127;
  int r = idx >> 7;          // b*2304 + pix
  int pix = r % 2304;
  int b = r / 2304;
  int y = pix / 48, x = pix % 48;
  const float* xb = X + b * 2304;
  float acc = B[c];
#pragma unroll
  for (int ky = 0; ky < 3; ++ky) {
    int iy = y + ky - 1;
    if (iy < 0 || iy >= 48) continue;
#pragma unroll
    for (int kx = 0; kx < 3; ++kx) {
      int ix = x + kx - 1;
      if (ix < 0 || ix >= 48) continue;
      acc += xb[iy * 48 + ix] * W[c * 9 + ky * 3 + kx];
    }
  }
  float sc = g[c] * rsqrtf(vr[c] + BN_EPS);
  H[(size_t)r * 128 + c] = (acc - mn[c]) * sc + be[c];
}

// ---- IF node #1 over T (same input each step) + 2x2 maxpool ---------------
__global__ __launch_bounds__(256) void if1_pool_k(
    const float* __restrict__ H, _Float16* __restrict__ P1) {
  int idx = blockIdx.x * 256 + threadIdx.x;
  if (idx >= BATCH * 576 * 128) return;
  int c = idx & 127;
  int r = idx >> 7;
  int pix = r % 576;
  int b = r / 576;
  int py = pix / 24, px = pix % 24;
  float h[4], v[4] = {0.f, 0.f, 0.f, 0.f};
#pragma unroll
  for (int j = 0; j < 4; ++j) {
    int iy = py * 2 + (j >> 1), ix = px * 2 + (j & 1);
    h[j] = H[((size_t)b * 2304 + iy * 48 + ix) * 128 + c];
  }
  for (int t = 0; t < T_STEPS; ++t) {
    float smax = 0.f;
#pragma unroll
    for (int j = 0; j < 4; ++j) {
      v[j] += h[j];
      if (v[j] >= 1.f) { smax = 1.f; v[j] = 0.f; }  // hard reset
    }
    P1[((size_t)(t * BATCH + b) * 576 + pix) * 128 + c] = (_Float16)smax;
  }
}

// ---- conv2 implicit GEMM via WMMA -----------------------------------------
// Block: 8 waves = 4 (pixel dim) x 2 (channel dim). Tile: 64 px x 128 ch.
// Halo tile (10x10x128 f16 NHWC) staged in LDS; B prepacked fragment-order.
__global__ __launch_bounds__(256) void conv2_wmma_k(
    const _Float16* __restrict__ P1, const _Float16* __restrict__ Wp,
    const float* __restrict__ Bf, float* __restrict__ C2) {
  __shared__ __align__(16) _Float16 Hs[100 * 128];
  int tid = threadIdx.x;
  int lane = tid & 31;
  int wave = tid >> 5;
  int waveM = wave & 3;
  int waveN = wave >> 2;
  int laneM = lane & 15;
  int laneHi = lane >> 4;
  int img = blockIdx.x;          // 0..255  (t*32+b)
  int tile = blockIdx.y;         // 0..8    (3x3 spatial tiles of 8x8 px)
  int ty = tile / 3, tx = tile % 3;

  const _Float16* base = P1 + (size_t)img * (576 * 128);
  for (int idx = tid; idx < 1600; idx += 256) {   // 12800 halfs / 8 per chunk
    int hp = idx >> 4;                            // halo pixel 0..99
    int c8 = (idx & 15) << 3;                     // channel chunk
    int hy = hp / 10, hx = hp % 10;
    int gy = ty * 8 - 1 + hy, gx = tx * 8 - 1 + hx;
    bool inb = (gy >= 0 && gy < 24 && gx >= 0 && gx < 24);
#if USE_ASYNC_STAGE
    if (inb) {
      ASYNC_B128(base + ((size_t)(gy * 24 + gx) << 7) + c8,
                 Hs + hp * 128 + c8);
    } else {
      v8h z = {};
      *(v8h*)(Hs + hp * 128 + c8) = z;            // zero-pad border
    }
#else
    v8h val = {};
    if (inb) val = *(const v8h*)(base + ((size_t)(gy * 24 + gx) << 7) + c8);
    *(v8h*)(Hs + hp * 128 + c8) = val;
#endif
  }
#if USE_ASYNC_STAGE
  ASYNC_WAIT();
#endif
  __syncthreads();

  v8f acc[4] = {};
  int p = waveM * 16 + laneM;          // output pixel within 8x8 tile
  int oy = p >> 3, ox = p & 7;
#pragma unroll
  for (int tap = 0; tap < 9; ++tap) {
    int ky = tap / 3, kx = tap % 3;
    const _Float16* arow = Hs + ((oy + ky) * 10 + (ox + kx)) * 128;
#pragma unroll
    for (int kc = 0; kc < 4; ++kc) {
      int koff = kc * 32 + laneHi * 8;
      v16h a = frag_cat(*(const v8h*)(arow + koff),
                        *(const v8h*)(arow + koff + 16));
#pragma unroll
      for (int f = 0; f < 4; ++f) {
        int nT = waveN * 4 + f;
        const _Float16* bp =
            Wp + (((size_t)(tap * 8 + nT) * 4 + kc) << 9) + lane * 16;
        v16h bfrag = frag_cat(*(const v8h*)(bp), *(const v8h*)(bp + 8));
        acc[f] = __builtin_amdgcn_wmma_f32_16x16x32_f16(
            false, a, false, bfrag, (short)0, acc[f], false, false);
      }
    }
  }
#pragma unroll
  for (int f = 0; f < 4; ++f) {
    int o = waveN * 64 + f * 16 + laneM;
    float bi = Bf[o];
#pragma unroll
    for (int r = 0; r < 8; ++r) {
      int prow = waveM * 16 + r + laneHi * 8;
      int yy = ty * 8 + (prow >> 3);
      int xx = tx * 8 + (prow & 7);
      C2[((size_t)img * 576 + yy * 24 + xx) * 128 + o] = acc[f][r] + bi;
    }
  }
}

// ---- IF node #2 over T + 2x2 maxpool; output NCHW f16 (flatten order) -----
__global__ __launch_bounds__(256) void if2_pool_k(
    const float* __restrict__ C2, _Float16* __restrict__ P2) {
  int idx = blockIdx.x * 256 + threadIdx.x;
  if (idx >= BATCH * 144 * 128) return;
  int c = idx & 127;
  int r = idx >> 7;
  int pix = r % 144;
  int b = r / 144;
  int py = pix / 12, px = pix % 12;
  float v[4] = {0.f, 0.f, 0.f, 0.f};
  for (int t = 0; t < T_STEPS; ++t) {
    int img = t * BATCH + b;
    float smax = 0.f;
#pragma unroll
    for (int j = 0; j < 4; ++j) {
      int iy = py * 2 + (j >> 1), ix = px * 2 + (j & 1);
      float x = C2[((size_t)img * 576 + iy * 24 + ix) * 128 + c];
      v[j] += x;
      if (v[j] >= 1.f) { smax = 1.f; v[j] = 0.f; }
    }
    P2[((size_t)img * 128 + c) * 144 + pix] = (_Float16)smax;
  }
}

// ---- generic WMMA GEMM: C[M,N] = A[M,K](f16) * Bpacked + bias -------------
// Block: 8 waves = 4(M) x 2(N); tile 64x128; A staged in LDS 64x128 per step.
__global__ __launch_bounds__(256) void gemm_f16_wmma_k(
    const _Float16* __restrict__ A, const _Float16* __restrict__ Bp,
    const float* __restrict__ bias, float* __restrict__ C,
    int M, int N, int K) {
  __shared__ __align__(16) _Float16 As[64 * 128];
  int tid = threadIdx.x;
  int lane = tid & 31;
  int wave = tid >> 5;
  int waveM = wave & 3;
  int waveN = wave >> 2;
  int laneM = lane & 15;
  int laneHi = lane >> 4;
  int m0 = blockIdx.y * 64;
  int n0 = blockIdx.x * 128;
  int kBlocks = K >> 5;
  v8f acc[4] = {};
  for (int kb0 = 0; kb0 < K; kb0 += 128) {
    __syncthreads();
    if (kb0 + 128 < K)   // hint next A panel into cache (global_prefetch)
      __builtin_prefetch(A + (size_t)(m0 + (tid >> 2)) * K + kb0 + 128, 0, 1);
#if USE_ASYNC_STAGE
    for (int idx = tid; idx < 1024; idx += 256) {  // 8192 halfs / 8 per chunk
      int row = idx >> 4;
      int c8 = (idx & 15) << 3;
      ASYNC_B128(A + (size_t)(m0 + row) * K + kb0 + c8,
                 As + row * 128 + c8);
    }
    ASYNC_WAIT();
#else
    for (int idx = tid; idx < 1024; idx += 256) {  // 8192 halfs / 8 per chunk
      int row = idx >> 4;
      int c8 = (idx & 15) << 3;
      *(v8h*)(As + row * 128 + c8) =
          *(const v8h*)(A + (size_t)(m0 + row) * K + kb0 + c8);
    }
#endif
    __syncthreads();
#pragma unroll
    for (int kc = 0; kc < 4; ++kc) {
      int row = waveM * 16 + laneM;
      int koff = kc * 32 + laneHi * 8;
      v16h a = frag_cat(*(const v8h*)(As + row * 128 + koff),
                        *(const v8h*)(As + row * 128 + koff + 16));
      int kBlock = (kb0 >> 5) + kc;
#pragma unroll
      for (int f = 0; f < 4; ++f) {
        int nT = (n0 >> 4) + waveN * 4 + f;
        const _Float16* bp =
            Bp + ((size_t)(nT * kBlocks + kBlock) << 9) + lane * 16;
        v16h bfrag = frag_cat(*(const v8h*)(bp), *(const v8h*)(bp + 8));
        acc[f] = __builtin_amdgcn_wmma_f32_16x16x32_f16(
            false, a, false, bfrag, (short)0, acc[f], false, false);
      }
    }
  }
#pragma unroll
  for (int f = 0; f < 4; ++f) {
    int col = n0 + waveN * 64 + f * 16 + laneM;
    float bi = bias[col];
#pragma unroll
    for (int r = 0; r < 8; ++r) {
      int row = m0 + waveM * 16 + r + laneHi * 8;
      C[(size_t)row * N + col] = acc[f][r] + bi;
    }
  }
}

// ---- LIF over T (tau=2, decay_input, hard reset); spikes out as f16 -------
__global__ __launch_bounds__(256) void lif_spikes_k(
    const float* __restrict__ X, _Float16* __restrict__ S, int R) {
  int idx = blockIdx.x * 256 + threadIdx.x;
  if (idx >= R) return;
  float v = 0.f;
  for (int t = 0; t < T_STEPS; ++t) {
    float x = X[(size_t)t * R + idx];
    v = 0.5f * (v + x);               // v + (x - v)/tau, tau = 2
    float s = 0.f;
    if (v >= 1.f) { s = 1.f; v = 0.f; }
    S[(size_t)t * R + idx] = (_Float16)s;
  }
}

// ---- fc3 (128->7) + LIF + mean over T -------------------------------------
__global__ __launch_bounds__(256) void fc3_lif_mean_k(
    const _Float16* __restrict__ L2, const float* __restrict__ W,
    const float* __restrict__ Bz, float* __restrict__ out) {
  int idx = blockIdx.x * 256 + threadIdx.x;
  if (idx >= BATCH * 7) return;
  int o = idx % 7;
  int b = idx / 7;
  float v = 0.f, acc = 0.f;
  for (int t = 0; t < T_STEPS; ++t) {
    const _Float16* l = L2 + (size_t)(t * BATCH + b) * 128;
    float x = Bz[o];
#pragma unroll 8
    for (int k = 0; k < 128; ++k) x += (float)l[k] * W[o * 128 + k];
    v = 0.5f * (v + x);
    if (v >= 1.f) { acc += 1.f; v = 0.f; }
  }
  out[b * 7 + o] = acc * 0.125f;
}

// ---------------------------------------------------------------------------
extern "C" void kernel_launch(void* const* d_in, const int* in_sizes, int n_in,
                              void* d_out, int out_size, void* d_ws, size_t ws_size,
                              hipStream_t stream) {
  (void)in_sizes; (void)n_in; (void)out_size; (void)ws_size;
  const float* x    = (const float*)d_in[0];
  const float* c1w  = (const float*)d_in[1];
  const float* c1b  = (const float*)d_in[2];
  const float* bn1g = (const float*)d_in[3];
  const float* bn1b = (const float*)d_in[4];
  const float* bn1m = (const float*)d_in[5];
  const float* bn1v = (const float*)d_in[6];
  const float* c2w  = (const float*)d_in[7];
  const float* c2b  = (const float*)d_in[8];
  const float* bn2g = (const float*)d_in[9];
  const float* bn2b = (const float*)d_in[10];
  const float* bn2m = (const float*)d_in[11];
  const float* bn2v = (const float*)d_in[12];
  const float* fc1w = (const float*)d_in[13];
  const float* fc1b = (const float*)d_in[14];
  const float* fc2w = (const float*)d_in[15];
  const float* fc2b = (const float*)d_in[16];
  const float* fc3w = (const float*)d_in[17];
  const float* fc3b = (const float*)d_in[18];

  // Workspace layout (bytes), total ~166 MB; regions A/B reused across stages.
  char* ws = (char*)d_ws;
  _Float16* W2P  = (_Float16*)(ws + 0);           //   294,912
  float*    B2F  = (float*)   (ws + 294912);      //       512
  _Float16* FC1P = (_Float16*)(ws + 295424);      // 42,467,328
  _Float16* FC2P = (_Float16*)(ws + 42762752);    //   294,912
  char* regA = ws + 43057664;                     // 75,497,472 (H then C2)
  char* regB = ws + 118555136;                    // 37,748,736 (P1 then fc outs)
  char* regC = ws + 156303872;                    //  9,437,184 (P2)
  float*    H    = (float*)regA;
  _Float16* P1   = (_Float16*)regB;
  float*    C2   = (float*)regA;                  // overwrites H (dead)
  _Float16* P2   = (_Float16*)regC;
  float*    FC1O = (float*)(regB + 0);            // overwrites P1 (dead)
  _Float16* L1   = (_Float16*)(regB + 2097152);
  float*    FC2O = (float*)(regB + 4194304);
  _Float16* L2   = (_Float16*)(regB + 6291456);
  float* out = (float*)d_out;

  // 1) weight prep / fragment packing
  pack_conv2_k<<<576, 256, 0, stream>>>(c2w, c2b, bn2g, bn2b, bn2m, bn2v, W2P, B2F);
  pack_fc_k<<<82944, 256, 0, stream>>>(fc1w, FC1P, 18432, 576, 21233664LL);
  pack_fc_k<<<576, 256, 0, stream>>>(fc2w, FC2P, 1152, 36, 147456LL);

  // 2) conv1 + BN1  -> H [32,48,48,128] f32 (NHWC)
  conv1_bn_k<<<36864, 256, 0, stream>>>(x, c1w, c1b, bn1g, bn1b, bn1m, bn1v, H);

  // 3) IF1 + pool -> P1 spikes [256,24,24,128] f16
  if1_pool_k<<<9216, 256, 0, stream>>>(H, P1);

  // 4) conv2 (+BN folded) implicit GEMM on WMMA -> C2 [256,24,24,128] f32
  conv2_wmma_k<<<dim3(256, 9), 256, 0, stream>>>(P1, W2P, B2F, C2);

  // 5) IF2 + pool -> P2 spikes [256,128,12,12] f16 (flatten order)
  if2_pool_k<<<2304, 256, 0, stream>>>(C2, P2);

  // 6) fc1 [256,18432]x[18432,1152] WMMA -> FC1O; LIF -> L1
  gemm_f16_wmma_k<<<dim3(9, 4), 256, 0, stream>>>(P2, FC1P, fc1b, FC1O, 256, 1152, 18432);
  lif_spikes_k<<<144, 256, 0, stream>>>(FC1O, L1, 36864);

  // 7) fc2 [256,1152]x[1152,128] WMMA -> FC2O; LIF -> L2
  gemm_f16_wmma_k<<<dim3(1, 4), 256, 0, stream>>>(L1, FC2P, fc2b, FC2O, 256, 128, 1152);
  lif_spikes_k<<<16, 256, 0, stream>>>(FC2O, L2, 4096);

  // 8) fc3 + LIF + mean over T -> out [32,7]
  fc3_lif_mean_k<<<1, 256, 0, stream>>>(L2, fc3w, fc3b, out);
}
